// MambaBlock_19885698581128
// MI455X (gfx1250) — matrix operand order repaired and checked by reading, not compiled, not run on hardware
//
#include <hip/hip_runtime.h>
#include <hip/hip_bf16.h>
#include <stdint.h>

#define D_MODEL 1024
#define D_STATE 16
#define SEQ     4096
#define BATCH   8
#define TOKENS  (BATCH * SEQ)   // 32768
#define NKSTEPS (D_MODEL / 32)  // 32

typedef __attribute__((ext_vector_type(16))) __bf16 v16bf;
typedef __attribute__((ext_vector_type(8)))  float  v8f;
typedef __attribute__((ext_vector_type(4)))  int    v4i;   // true clang vector (HIP int4 is a struct)

#define AS_GLOBAL __attribute__((address_space(1)))
#define AS_LDS    __attribute__((address_space(3)))

#if __has_builtin(__builtin_amdgcn_global_load_async_to_lds_b128) && \
    __has_builtin(__builtin_amdgcn_s_wait_asynccnt)
#define USE_ASYNC_LDS 1
#else
#define USE_ASYNC_LDS 0
#endif

union Frag16 {
    v16bf v;
    uint4 q[2];
    __bf16 h[16];
    unsigned short s[16];
};

// Native f32 -> bf16 (RNE) convert; lowers to v_cvt_pk_bf16_f32 on gfx1250.
__device__ __forceinline__ unsigned short f2bf(float f) {
    union { __bf16 b; unsigned short s; } u;
    u.b = (__bf16)f;
    return u.s;
}

// ---------------------------------------------------------------------------
// Kernel 1: LayerNorm -> xn (bf16).  One block per token, 256 threads, 4 el/thr.
// ---------------------------------------------------------------------------
__global__ __launch_bounds__(256)
void ln_kernel(const float* __restrict__ x,
               const float* __restrict__ lnw,
               const float* __restrict__ lnb,
               unsigned short* __restrict__ xnbf) {
    const int t   = blockIdx.x;
    const int tid = threadIdx.x;
    const float* row = x + (size_t)t * D_MODEL;

    float v[4];
    float s = 0.f;
#pragma unroll
    for (int i = 0; i < 4; ++i) { v[i] = row[tid + 256 * i]; s += v[i]; }

    __shared__ float red[256];
    red[tid] = s;
    __syncthreads();
    for (int off = 128; off > 0; off >>= 1) {
        if (tid < off) red[tid] += red[tid + off];
        __syncthreads();
    }
    const float mu = red[0] * (1.0f / (float)D_MODEL);
    __syncthreads();

    float vs = 0.f;
#pragma unroll
    for (int i = 0; i < 4; ++i) { float d = v[i] - mu; vs += d * d; }
    red[tid] = vs;
    __syncthreads();
    for (int off = 128; off > 0; off >>= 1) {
        if (tid < off) red[tid] += red[tid + off];
        __syncthreads();
    }
    const float var  = red[0] * (1.0f / (float)D_MODEL);
    const float rstd = rsqrtf(var + 1e-5f);

#pragma unroll
    for (int i = 0; i < 4; ++i) {
        const int d = tid + 256 * i;
        const float xn = (v[i] - mu) * rstd * lnw[d] + lnb[d];
        xnbf[(size_t)t * D_MODEL + d] = f2bf(xn);
    }
}

// ---------------------------------------------------------------------------
// Kernel 2: u = xn @ (Ws+Wi)^T + bs + bi   -> [T, 16] f32
// One wave per 16-token tile; K-loop of v_wmma_f32_16x16x32_bf16 over D.
// ---------------------------------------------------------------------------
__global__ __launch_bounds__(32)
void uproj_kernel(const unsigned short* __restrict__ xnbf,
                  const float* __restrict__ Ws,
                  const float* __restrict__ bs,
                  const float* __restrict__ Wi,
                  const float* __restrict__ bi,
                  float* __restrict__ u) {
    const int t0      = blockIdx.x * 16;
    const int lane    = threadIdx.x;
    const int halfsel = lane >> 4;      // 0 or 1
    const int l15     = lane & 15;
    const int koffA   = halfsel * 16;   // A: 16 consecutive K starting at 0 / 16
    const int koffB   = halfsel * 16;   // B: 16 consecutive K starting at 0 / 16

    v8f acc = {0.f, 0.f, 0.f, 0.f, 0.f, 0.f, 0.f, 0.f};

    for (int k0 = 0; k0 < D_MODEL; k0 += 32) {
        Frag16 a, b;
        const unsigned short* ap = xnbf + (size_t)(t0 + l15) * D_MODEL + k0;
        a.q[0] = *(const uint4*)(ap + koffA);
        a.q[1] = *(const uint4*)(ap + koffA + 8);

        const float* wsrow = Ws + (size_t)l15 * D_MODEL + k0 + koffB;
        const float* wirow = Wi + (size_t)l15 * D_MODEL + k0 + koffB;
#pragma unroll
        for (int j = 0; j < 16; ++j) b.h[j] = (__bf16)(wsrow[j] + wirow[j]);

        acc = __builtin_amdgcn_wmma_f32_16x16x32_bf16(
            false, a.v, false, b.v, (short)0, acc, false, false);
    }

    const float bias = bs[l15] + bi[l15];
#pragma unroll
    for (int r = 0; r < 8; ++r) {
        const int t = t0 + r + 8 * halfsel;
        u[(size_t)t * D_STATE + l15] = acc[r] + bias;
    }
}

// ---------------------------------------------------------------------------
// Kernel 3: sequential recurrence h_t = tanh(h A^T + u_t), per batch.
// 1 wave per batch; lane n holds h[n]; __shfl broadcast (wave32).
// Writes H as bf16 for the output-projection WMMA.
// ---------------------------------------------------------------------------
__global__ __launch_bounds__(32)
void scan_kernel(const float* __restrict__ u,
                 const float* __restrict__ A,
                 unsigned short* __restrict__ Hbf) {
    const int b    = blockIdx.x;
    const int lane = threadIdx.x;
    const int n    = lane & 15;

    float arow[16];
#pragma unroll
    for (int m = 0; m < 16; ++m) arow[m] = A[n * 16 + m];

    const float* ub = u + (size_t)b * SEQ * D_STATE;
    unsigned short* hb = Hbf + (size_t)b * SEQ * D_STATE;

    float h = 0.f;
    for (int s = 0; s < SEQ; ++s) {
        float acc = ub[s * D_STATE + n];
#pragma unroll
        for (int m = 0; m < 16; ++m) {
            const float hm = __shfl(h, m, 32);
            acc = fmaf(arow[m], hm, acc);
        }
        h = tanhf(acc);
        if (lane < 16) hb[s * D_STATE + n] = f2bf(h);
    }
}

// ---------------------------------------------------------------------------
// Kernel 4: fused gate GEMM + out projection + sigmoid gate + residual.
// Block tile 128 tokens x 128 cols; 8 waves (256 thr).
// Double-buffered k-pipeline: A tile staged global->LDS with
// GLOBAL_LOAD_ASYNC_TO_LDS_B128 (ASYNCcnt) for step k+1 while step k's
// 8 WMMAs execute; B tile (Wg) converted f32->bf16 through VGPRs.
// Each wave owns 2x4 subtiles of 16x16; epilogue adds H@Wo^T (K padded),
// sigmoid gate and residual.
// ---------------------------------------------------------------------------
__global__ __launch_bounds__(256)
void fused_out_kernel(const unsigned short* __restrict__ xnbf,
                      const unsigned short* __restrict__ Hbf,
                      const float* __restrict__ Wg,
                      const float* __restrict__ bg,
                      const float* __restrict__ Wo,
                      const float* __restrict__ bo,
                      const float* __restrict__ x,
                      float* __restrict__ out) {
    const int trow0 = blockIdx.x * 128;
    const int e0    = blockIdx.y * 128;
    const int tid   = threadIdx.x;
    const int wave  = tid >> 5;
    const int lane  = tid & 31;
    const int wm    = wave >> 1;        // 0..3  (row group of 32 tokens)
    const int wn    = wave & 1;         // 0..1  (col group of 64 cols)
    const int halfsel = lane >> 4;
    const int l15     = lane & 15;
    const int koffA   = halfsel * 8;
    const int koffB   = halfsel * 16;

    __shared__ unsigned short As[2][128 * 32];  // [buf][row][k]  2 x 8KB
    __shared__ unsigned short Bs[2][128 * 32];  // [buf][col][k]  2 x 8KB

    v8f acc[2][4];
#pragma unroll
    for (int i = 0; i < 2; ++i)
#pragma unroll
        for (int j = 0; j < 4; ++j)
            acc[i][j] = (v8f){0.f, 0.f, 0.f, 0.f, 0.f, 0.f, 0.f, 0.f};

    const int ldrow = tid >> 1;         // 0..127
    const int ldkc  = (tid & 1) * 16;   // 0 or 16
    const unsigned short* agbase = xnbf + (size_t)(trow0 + ldrow) * D_MODEL + ldkc;
    const float*          bgbase = Wg   + (size_t)(e0    + ldrow) * D_MODEL + ldkc;

    // ---- prologue: stage k-step 0 into buffer 0 ----
    {
        const unsigned short* gp = agbase;
#if USE_ASYNC_LDS
        __builtin_amdgcn_global_load_async_to_lds_b128(
            (AS_GLOBAL v4i*)gp, (AS_LDS v4i*)&As[0][ldrow * 32 + ldkc], 0, 0);
        __builtin_amdgcn_global_load_async_to_lds_b128(
            (AS_GLOBAL v4i*)(gp + 8), (AS_LDS v4i*)&As[0][ldrow * 32 + ldkc + 8], 0, 0);
#else
        *(uint4*)&As[0][ldrow * 32 + ldkc]     = *(const uint4*)gp;
        *(uint4*)&As[0][ldrow * 32 + ldkc + 8] = *(const uint4*)(gp + 8);
#endif
        Frag16 tmp;
#pragma unroll
        for (int j = 0; j < 16; ++j) tmp.h[j] = (__bf16)bgbase[j];
        *(uint4*)&Bs[0][ldrow * 32 + ldkc]     = tmp.q[0];
        *(uint4*)&Bs[0][ldrow * 32 + ldkc + 8] = tmp.q[1];
    }
#if USE_ASYNC_LDS
    __builtin_amdgcn_s_wait_asynccnt(0);
#endif
    __syncthreads();

    for (int kt = 0; kt < NKSTEPS; ++kt) {
        const int cur = kt & 1;
        const int nxt = cur ^ 1;

        // ---- stage k-step kt+1 into the other buffer (overlaps WMMAs) ----
        if (kt + 1 < NKSTEPS) {
            const int k1 = (kt + 1) * 32;
            const unsigned short* gp = agbase + k1;
#if USE_ASYNC_LDS
            __builtin_amdgcn_global_load_async_to_lds_b128(
                (AS_GLOBAL v4i*)gp, (AS_LDS v4i*)&As[nxt][ldrow * 32 + ldkc], 0, 0);
            __builtin_amdgcn_global_load_async_to_lds_b128(
                (AS_GLOBAL v4i*)(gp + 8), (AS_LDS v4i*)&As[nxt][ldrow * 32 + ldkc + 8], 0, 0);
#else
            *(uint4*)&As[nxt][ldrow * 32 + ldkc]     = *(const uint4*)gp;
            *(uint4*)&As[nxt][ldrow * 32 + ldkc + 8] = *(const uint4*)(gp + 8);
#endif
            const float* wgp = bgbase + k1;
            Frag16 tmp;
#pragma unroll
            for (int j = 0; j < 16; ++j) tmp.h[j] = (__bf16)wgp[j];
            *(uint4*)&Bs[nxt][ldrow * 32 + ldkc]     = tmp.q[0];
            *(uint4*)&Bs[nxt][ldrow * 32 + ldkc + 8] = tmp.q[1];
        }

        // ---- compute from current buffer ----
        Frag16 af[2], bf[4];
#pragma unroll
        for (int sr = 0; sr < 2; ++sr) {
            const int rowl = wm * 32 + sr * 16 + l15;
            af[sr].q[0] = *(const uint4*)&As[cur][rowl * 32 + koffB];
            af[sr].q[1] = *(const uint4*)&As[cur][rowl * 32 + koffB + 8];
        }
#pragma unroll
        for (int sc = 0; sc < 4; ++sc) {
            const int coll = wn * 64 + sc * 16 + l15;
            bf[sc].q[0] = *(const uint4*)&Bs[cur][coll * 32 + koffB];
            bf[sc].q[1] = *(const uint4*)&Bs[cur][coll * 32 + koffB + 8];
        }
#pragma unroll
        for (int sr = 0; sr < 2; ++sr)
#pragma unroll
            for (int sc = 0; sc < 4; ++sc)
                acc[sr][sc] = __builtin_amdgcn_wmma_f32_16x16x32_bf16(
                    false, af[sr].v, false, bf[sc].v, (short)0,
                    acc[sr][sc], false, false);

        // ---- buffer swap: next tile landed + everyone done reading cur ----
#if USE_ASYNC_LDS
        __builtin_amdgcn_s_wait_asynccnt(0);
#endif
        __syncthreads();
    }

    // ---- epilogue: out = H @ Wo^T (K=16 padded to 32), gate, residual ----
    Frag16 hf[2];
#pragma unroll
    for (int sr = 0; sr < 2; ++sr) {
        const int t = trow0 + wm * 32 + sr * 16 + l15;
        const unsigned short* hp = Hbf + (size_t)t * D_STATE;
        hf[sr].q[0] = halfsel == 0 ? *(const uint4*)hp : (uint4){0u, 0u, 0u, 0u};       // K 0..7 (real) / 16..23 (zero pad)
        hf[sr].q[1] = halfsel == 0 ? *(const uint4*)(hp + 8) : (uint4){0u, 0u, 0u, 0u}; // K 8..15 (real) / 24..31 (zero pad)
    }
    Frag16 wf[4];
#pragma unroll
    for (int sc = 0; sc < 4; ++sc) {
        const int e = e0 + wn * 64 + sc * 16 + l15;
        if (halfsel == 0) {
            const float* wp = Wo + (size_t)e * D_STATE;
#pragma unroll
            for (int j = 0; j < 16; ++j) wf[sc].h[j] = (__bf16)wp[j]; // K 0..15
        } else {
            wf[sc].q[0] = (uint4){0u, 0u, 0u, 0u};                    // K 16..31
            wf[sc].q[1] = (uint4){0u, 0u, 0u, 0u};
        }
    }

#pragma unroll
    for (int sr = 0; sr < 2; ++sr) {
#pragma unroll
        for (int sc = 0; sc < 4; ++sc) {
            v8f oacc = (v8f){0.f, 0.f, 0.f, 0.f, 0.f, 0.f, 0.f, 0.f};
            oacc = __builtin_amdgcn_wmma_f32_16x16x32_bf16(
                false, hf[sr].v, false, wf[sc].v, (short)0, oacc, false, false);

            const int col = e0 + wn * 64 + sc * 16 + l15;
            const float bgc = bg[col];
            const float boc = bo[col];
#pragma unroll
            for (int r = 0; r < 8; ++r) {
                const int t = trow0 + wm * 32 + sr * 16 + r + 8 * halfsel;
                const float g  = acc[sr][sc][r] + bgc;
                const float sg = 1.0f / (1.0f + expf(-g));
                const float ov = oacc[r] + boc;
                out[(size_t)t * D_MODEL + col] =
                    ov * sg + x[(size_t)t * D_MODEL + col];
            }
        }
    }
}

// ---------------------------------------------------------------------------
extern "C" void kernel_launch(void* const* d_in, const int* in_sizes, int n_in,
                              void* d_out, int out_size, void* d_ws, size_t ws_size,
                              hipStream_t stream) {
    const float* x    = (const float*)d_in[0];
    const float* Ws   = (const float*)d_in[1];
    const float* bs   = (const float*)d_in[2];
    const float* Wi   = (const float*)d_in[3];
    const float* bi   = (const float*)d_in[4];
    const float* Wo   = (const float*)d_in[5];
    const float* bo   = (const float*)d_in[6];
    const float* Wg   = (const float*)d_in[7];
    const float* bg   = (const float*)d_in[8];
    const float* lnw  = (const float*)d_in[9];
    const float* lnb  = (const float*)d_in[10];
    const float* A    = (const float*)d_in[11];
    float* out = (float*)d_out;

    // workspace layout
    unsigned short* xnbf = (unsigned short*)d_ws;                           // 64 MB
    unsigned short* Hbf  = (unsigned short*)((char*)d_ws + (size_t)TOKENS * D_MODEL * 2);
    float*          u    = (float*)((char*)Hbf + (size_t)TOKENS * D_STATE * 2);

    ln_kernel<<<TOKENS, 256, 0, stream>>>(x, lnw, lnb, xnbf);
    uproj_kernel<<<TOKENS / 16, 32, 0, stream>>>(xnbf, Ws, bs, Wi, bi, u);
    scan_kernel<<<BATCH, 32, 0, stream>>>(u, A, Hbf);
    fused_out_kernel<<<dim3(TOKENS / 128, D_MODEL / 128), 256, 0, stream>>>(
        xnbf, Hbf, Wg, bg, Wo, bo, x, out);
}
